// AdjLayer_23776938950933
// MI455X (gfx1250) — matrix operand order repaired and checked by reading, not compiled, not run on hardware
//
#include <hip/hip_runtime.h>
#include <hip/hip_bf16.h>
#include <math.h>

typedef __attribute__((ext_vector_type(16))) _Float16 v16h;
typedef __attribute__((ext_vector_type(8)))  float    v8f;

#define NF     156          // clusters / conv filters
#define NPAD   160          // padded to 10 x 16 N-tiles
#define NT     10           // number of 16-wide N tiles
#define KPAD   32           // K = 27 (3*3*3) padded to one WMMA K-step
#define OUTW   220
#define INW    222
#define MTILES 3025         // 48400 / 16 exactly
#define ADJ_OFF (NF * NF)   // new_adj comes first in d_out, Sout after
#define WELEMS (27 * NF)    // 4212 raw f32 weights
#define WCHUNKS (WELEMS / 4) // 1053 16-byte chunks (4212 dwords, exact)

// Fused: async-staged weights -> implicit-GEMM conv (WMMA f16->f32) + bias +
// row softmax + non-temporal Sout stream + column-sum accumulation.
__global__ __launch_bounds__(128)
void conv_softmax_kernel(const float* __restrict__ x,   // [1,222,222,3]
                         const float* __restrict__ W,   // [3,3,3,156] HWIO
                         const float* __restrict__ b,   // [156]
                         float* __restrict__ out,       // d_out
                         float* __restrict__ csum)      // ws column sums [NPAD]
{
    __shared__ float    wstage[WELEMS];     // raw f32 weights, async-DMA'd
    __shared__ _Float16 w16[NPAD * KPAD];   // weights as [n][k], f16, zero-padded
    __shared__ float    cblk[NPAD];         // per-block column-sum accumulator

    const int tid = threadIdx.x;

    // ---- async bulk copy of W into LDS (ASYNCcnt path, B128 chunks) ----
    for (int c = tid; c < WCHUNKS; c += 128) {
        const unsigned lds_dst =
            (unsigned)(uintptr_t)&wstage[c * 4];            // low 32b = LDS offset
        const unsigned long long gaddr =
            (unsigned long long)(uintptr_t)(W + c * 4);
        asm volatile("global_load_async_to_lds_b128 %0, %1, off"
                     :: "v"(lds_dst), "v"(gaddr) : "memory");
    }
#if __has_builtin(__builtin_amdgcn_s_wait_asynccnt)
    __builtin_amdgcn_s_wait_asynccnt(0);
#else
    asm volatile("s_wait_asynccnt 0x0" ::: "memory");
#endif
    for (int i = tid; i < NPAD; i += 128) cblk[i] = 0.0f;
    __syncthreads();

    // ---- convert f32 -> f16 into WMMA-friendly [n][k] layout, zero-padded ----
    for (int i = tid; i < NPAD * KPAD; i += 128) {
        const int n = i >> 5, k = i & 31;
        const float v = (n < NF && k < 27) ? wstage[k * NF + n] : 0.0f;
        w16[i] = (_Float16)v;
    }
    __syncthreads();

    const int wave = tid >> 5;
    const int lane = tid & 31;
    const int tile = blockIdx.x * 4 + wave;   // one wave per 16-row tile (uniform)

    if (tile < MTILES) {
        // ---- build A fragment: 16x32 f16 im2col tile (doc'd A layout) ----
        const int m  = lane & 15;             // M = row within tile
        const int r  = tile * 16 + m;         // global output-pixel row
        const int oy = r / OUTW, ox = r % OUTW;
        const int kbase = (lane < 16) ? 0 : 8;

        __builtin_prefetch(&x[(oy * INW + ox) * 3], 0, 3);  // global_prefetch_b8

        v16h a;
        #pragma unroll
        for (int j = 0; j < 16; ++j) {
            const int k = kbase + j + ((j >= 8) ? 8 : 0);   // K index per layout
            float v = 0.0f;
            if (k < 27) {
                const int ky = k / 9, kx = (k % 9) / 3, ci = k % 3;
                v = x[((oy + ky) * INW + (ox + kx)) * 3 + ci];
            }
            a[j] = (_Float16)v;
        }

        // ---- 10 WMMAs: full K (27<=32) in a single K-step each ----
        const int khalf = (lane < 16) ? 0 : 8;    // B layout: lanes 0-15 K=0..7,16..23
        const int ncol  = lane & 15;              // N within tile

        v8f acc[NT];
        #pragma unroll
        for (int t = 0; t < NT; ++t) {
            const int n = t * 16 + ncol;
            typedef __attribute__((ext_vector_type(8))) _Float16 v8h_t;
            const v8h_t blo = *(const v8h_t*)&w16[n * KPAD + khalf];
            const v8h_t bhi = *(const v8h_t*)&w16[n * KPAD + 16 + khalf];
            const v16h bf = __builtin_shufflevector(blo, bhi, 0,1,2,3,4,5,6,7,8,9,10,11,12,13,14,15); // 2x ds_load_b128
            v8f c = {};
            acc[t] = __builtin_amdgcn_wmma_f32_16x16x32_f16(
                false, a, false, bf, (short)0, c, false, false);
        }

        float bias[NT];
        #pragma unroll
        for (int t = 0; t < NT; ++t) {
            const int n = t * 16 + ncol;
            bias[t] = (n < NF) ? b[n] : 0.0f;
        }

        float* Sout = out + ADJ_OFF;
        float colp[NT];
        #pragma unroll
        for (int t = 0; t < NT; ++t) colp[t] = 0.0f;

        // ---- row softmax: C-layout row lives in one VGPR across 16 lanes ----
        #pragma unroll
        for (int v = 0; v < 8; ++v) {
            const int row = tile * 16 + v + ((lane < 16) ? 0 : 8);
            float val[NT];
            float mx = -INFINITY;
            #pragma unroll
            for (int t = 0; t < NT; ++t) {
                const int n = t * 16 + ncol;
                val[t] = (n < NF) ? (acc[t][v] + bias[t]) : -INFINITY;
                mx = fmaxf(mx, val[t]);
            }
            #pragma unroll
            for (int s = 8; s >= 1; s >>= 1)        // half-wave max reduce
                mx = fmaxf(mx, __shfl_xor(mx, s, 32));
            float sum = 0.0f;
            #pragma unroll
            for (int t = 0; t < NT; ++t) {
                val[t] = __expf(val[t] - mx);       // pad cols -> exp(-inf)=0
                sum += val[t];
            }
            #pragma unroll
            for (int s = 8; s >= 1; s >>= 1)        // half-wave sum reduce
                sum += __shfl_xor(sum, s, 32);
            const float inv = 1.0f / sum;
            #pragma unroll
            for (int t = 0; t < NT; ++t) {
                const int n = t * 16 + ncol;
                const float p = val[t] * inv;
                if (n < NF) {
                    // 30 MB one-shot stream: non-temporal (TH_NT) store
                    __builtin_nontemporal_store(p, &Sout[row * NF + n]);
                    colp[t] += p;
                }
            }
        }

        // ---- column partial sums into LDS (ds_add_f32) ----
        #pragma unroll
        for (int t = 0; t < NT; ++t) {
            const int n = t * 16 + ncol;
            if (n < NF) atomicAdd(&cblk[n], colp[t]);
        }
    }
    __syncthreads();
    if (tid < NF) atomicAdd(&csum[tid], cblk[tid]);  // one global atomic / col / block
}

__global__ void zero_ws_kernel(float* __restrict__ csum) {
    if (threadIdx.x < NPAD) csum[threadIdx.x] = 0.0f;
}

__global__ void outer_kernel(const float* __restrict__ csum,
                             float* __restrict__ out) {
    const int i = blockIdx.x * 256 + threadIdx.x;
    if (i < NF * NF) {
        const int r = i / NF, c = i % NF;
        out[i] = csum[r] * csum[c];   // new_adj = outer(c, c)
    }
}

extern "C" void kernel_launch(void* const* d_in, const int* in_sizes, int n_in,
                              void* d_out, int out_size, void* d_ws, size_t ws_size,
                              hipStream_t stream) {
    const float* x = (const float*)d_in[0];   // [1,222,222,3] f32
    const float* W = (const float*)d_in[1];   // [3,3,3,156]  f32
    const float* b = (const float*)d_in[2];   // [156]        f32
    float* out  = (float*)d_out;              // [156*156] new_adj ++ [48400*156] Sout
    float* csum = (float*)d_ws;               // [160] column-sum accumulator

    zero_ws_kernel<<<1, 256, 0, stream>>>(csum);
    const int blocks = (MTILES + 3) / 4;      // 4 waves (tiles) per 128-thread block
    conv_softmax_kernel<<<blocks, 128, 0, stream>>>(x, W, b, out, csum);
    outer_kernel<<<(NF * NF + 255) / 256, 256, 0, stream>>>(csum, out);
}
